// GraphSAGE_25598005084435
// MI455X (gfx1250) — matrix operand order repaired and checked by reading, not compiled, not run on hardware
//
#include <hip/hip_runtime.h>
#include <hip/hip_bf16.h>

// ---------------------------------------------------------------------------
// Types for CDNA5 WMMA (wave32): v_wmma_f32_16x16x32_bf16
// ---------------------------------------------------------------------------
typedef __attribute__((ext_vector_type(16))) __bf16 v16bf;
typedef __attribute__((ext_vector_type(8)))  float  v8f;

// fp32 -> bf16 round-to-nearest-even (used in pre-passes and the bf16 GEMM epilogue)
__device__ __forceinline__ unsigned short f2bfbits(float f) {
    unsigned u = __float_as_uint(f);
    return (unsigned short)((u + 0x7FFFu + ((u >> 16) & 1u)) >> 16);
}
__device__ __forceinline__ float bf2f(unsigned short s) {
    return __uint_as_float(((unsigned)s) << 16);
}

// Bit-cast two 16-byte chunks into a v16bf fragment (register moves only)
__device__ __forceinline__ v16bf make_frag(uint4 lo, uint4 hi) {
    union { v16bf bf; uint4 q[2]; } u;
    u.q[0] = lo; u.q[1] = hi;
    return u.bf;
}

// ---------------------------------------------------------------------------
// Zero-fill (grid-stride)
// ---------------------------------------------------------------------------
__global__ void zero_f32_kernel(float* __restrict__ p, long long n) {
    long long i = (long long)blockIdx.x * blockDim.x + threadIdx.x;
    long long stride = (long long)gridDim.x * blockDim.x;
    for (; i < n; i += stride) p[i] = 0.0f;
}

// ---------------------------------------------------------------------------
// fp32 -> bf16 conversion pass (grid-stride, memory bound)
// ---------------------------------------------------------------------------
__global__ void cvt_bf16_kernel(const float* __restrict__ in,
                                unsigned short* __restrict__ out, long long n) {
    long long i = (long long)blockIdx.x * blockDim.x + threadIdx.x;
    long long stride = (long long)gridDim.x * blockDim.x;
    for (; i < n; i += stride) out[i] = f2bfbits(in[i]);
}

// ---------------------------------------------------------------------------
// h_neigh = bf16( agg / max(deg,1) )   (grid-stride; K = 128 -> row = i>>7)
// ---------------------------------------------------------------------------
__global__ void neigh_cvt_kernel(const float* __restrict__ agg,
                                 const float* __restrict__ deg,
                                 unsigned short* __restrict__ out, long long n) {
    long long i = (long long)blockIdx.x * blockDim.x + threadIdx.x;
    long long stride = (long long)gridDim.x * blockDim.x;
    for (; i < n; i += stride) {
        const long long row = i >> 7;
        const float inv = 1.0f / fmaxf(deg[row], 1.0f);
        out[i] = f2bfbits(agg[i] * inv);
    }
}

// ---------------------------------------------------------------------------
// Edge scatter: agg[dst] += bf16->f32(h[src]); deg[dst] += 1 (layer 1 only).
// 128 threads = 128 features per edge; fp32 atomics into L2-resident agg.
// ---------------------------------------------------------------------------
template<int K>
__global__ __launch_bounds__(K)
void sage_scatter_kernel(const unsigned short* __restrict__ h,
                         const int*  __restrict__ src,
                         const int*  __restrict__ dst,
                         float* __restrict__ agg,
                         float* __restrict__ deg,   // may be null
                         int nEdges) {
    const int t = threadIdx.x;
    for (int e = blockIdx.x; e < nEdges; e += gridDim.x) {
        const int s = src[e];
        const int d = dst[e];
        const float v = bf2f(h[(size_t)s * K + t]);
        atomicAdd(&agg[(size_t)d * K + t], v);
        if (deg != nullptr && t == 0) atomicAdd(&deg[d], 1.0f);
    }
}

// ---------------------------------------------------------------------------
// Fused SAGEConv dense kernel, all operands pre-converted to bf16:
//   out = [relu]( A_self @ Ws^T + A_neigh @ Wn^T + bias )
// One wave per 16x16 tile; hot loop = b128 loads + v_wmma only.
//
// Per-lane fragment addressing (CDNA5 ISA 7.12.2, wave32):
//   A: lane -> row M = lane%16, half = lane/16; lane data = two contiguous
//      8-element runs at K = kt + half*8 and K = kt + 16 + half*8.
//   B: lane -> col N = lane%16, half = lane/16; lane data = two contiguous
//      8-element runs at K = kt + half*8 and K = kt + 16 + half*8.
//   C/D: VGPR r -> row M = r + 8*half, col N = lane%16.
// ---------------------------------------------------------------------------
template<int K_DIM, int N_OUT, bool RELU, bool OUT_BF16>
__global__ __launch_bounds__((N_OUT / 16) * 32)
void sage_gemm_kernel(const unsigned short* __restrict__ A_self,   // [N,K] bf16
                      const unsigned short* __restrict__ A_neigh,  // [N,K] bf16
                      const unsigned short* __restrict__ Ws,       // [N_OUT,K] bf16
                      const unsigned short* __restrict__ Wn,       // [N_OUT,K] bf16
                      const float* __restrict__ bias,              // [N_OUT]
                      void* __restrict__ out_raw,                  // [N,N_OUT]
                      int nNodes) {
    const int wave = threadIdx.x >> 5;
    const int lane = threadIdx.x & 31;
    const int m    = lane & 15;
    const int half = lane >> 4;
    const int rowBase = blockIdx.x * 16;
    const int colBase = wave * 16;
    const int col = colBase + m;

    int row = rowBase + m;
    if (row >= nNodes) row = nNodes - 1;      // clamp loads; EXEC stays all-1s
    const unsigned short* aSelfRow  = A_self  + (size_t)row * K_DIM;
    const unsigned short* aNeighRow = A_neigh + (size_t)row * K_DIM;
    const unsigned short* wsCol = Ws + (size_t)col * K_DIM;
    const unsigned short* wnCol = Wn + (size_t)col * K_DIM;

    v8f c = {};

    // ---- self term ----
#pragma unroll
    for (int kt = 0; kt < K_DIM; kt += 32) {
        const int aOff = kt + half * 8;
        const v16bf a = make_frag(*(const uint4*)(aSelfRow + aOff),
                                  *(const uint4*)(aSelfRow + aOff + 16));
        const int bOff = kt + half * 8;
        const v16bf b = make_frag(*(const uint4*)(wsCol + bOff),
                                  *(const uint4*)(wsCol + bOff + 16));
        c = __builtin_amdgcn_wmma_f32_16x16x32_bf16(
                false, a, false, b, (short)0, c, false, false);
    }

    // ---- neighbor term ----
#pragma unroll
    for (int kt = 0; kt < K_DIM; kt += 32) {
        const int aOff = kt + half * 8;
        const v16bf a = make_frag(*(const uint4*)(aNeighRow + aOff),
                                  *(const uint4*)(aNeighRow + aOff + 16));
        const int bOff = kt + half * 8;
        const v16bf b = make_frag(*(const uint4*)(wnCol + bOff),
                                  *(const uint4*)(wnCol + bOff + 16));
        c = __builtin_amdgcn_wmma_f32_16x16x32_bf16(
                false, a, false, b, (short)0, c, false, false);
    }

    // ---- epilogue: bias (+ReLU) and store ----
    const float bc = bias[col];
#pragma unroll
    for (int r = 0; r < 8; r++) {
        const int orow = rowBase + (half << 3) + r;
        float v = c[r] + bc;
        if (RELU) v = fmaxf(v, 0.0f);
        if (orow < nNodes) {
            if (OUT_BF16)
                ((unsigned short*)out_raw)[(size_t)orow * N_OUT + col] = f2bfbits(v);
            else
                ((float*)out_raw)[(size_t)orow * N_OUT + col] = v;
        }
    }
}

// ---------------------------------------------------------------------------
// Launch sequence
// ---------------------------------------------------------------------------
extern "C" void kernel_launch(void* const* d_in, const int* in_sizes, int n_in,
                              void* d_out, int out_size, void* d_ws, size_t ws_size,
                              hipStream_t stream) {
    (void)n_in; (void)out_size; (void)ws_size;

    const float* x    = (const float*)d_in[0];
    const int*   esrc = (const int*)d_in[1];
    const int*   edst = (const int*)d_in[2];
    const float* Ws1  = (const float*)d_in[3];
    const float* Wn1  = (const float*)d_in[4];
    const float* b1   = (const float*)d_in[5];
    const float* Ws2  = (const float*)d_in[6];
    const float* Wn2  = (const float*)d_in[7];
    const float* b2   = (const float*)d_in[8];
    float* out = (float*)d_out;

    constexpr int IN  = 128;
    constexpr int HID = 128;
    constexpr int OUT = 64;
    const int N = in_sizes[0] / IN;
    const int E = in_sizes[1];

    // Workspace layout (16B-aligned sections):
    //   agg    [N*128 f32]
    //   deg    [N f32]
    //   h1     [N*128 bf16]
    //   hneigh [N*128 bf16]
    //   xbf    [N*128 bf16]
    //   wbf    [2*128*128 + 2*64*128 bf16]
    float* agg = (float*)d_ws;
    float* deg = agg + (size_t)N * IN;
    unsigned short* h1     = (unsigned short*)(deg + ((N + 3) & ~3));
    unsigned short* hneigh = h1 + (size_t)N * HID;
    unsigned short* xbf    = hneigh + (size_t)N * IN;
    unsigned short* ws1bf  = xbf + (size_t)N * IN;
    unsigned short* wn1bf  = ws1bf + HID * IN;
    unsigned short* ws2bf  = wn1bf + HID * IN;
    unsigned short* wn2bf  = ws2bf + OUT * HID;

    const int rowTiles = (N + 15) / 16;
    const int sgrid = (E < 1 << 19) ? E : (1 << 19);

    // ---- pre-convert operands to bf16 ----
    cvt_bf16_kernel<<<2048, 256, 0, stream>>>(x,   xbf,   (long long)N * IN);
    cvt_bf16_kernel<<<64, 256, 0, stream>>>(Ws1, ws1bf, (long long)HID * IN);
    cvt_bf16_kernel<<<64, 256, 0, stream>>>(Wn1, wn1bf, (long long)HID * IN);
    cvt_bf16_kernel<<<32, 256, 0, stream>>>(Ws2, ws2bf, (long long)OUT * HID);
    cvt_bf16_kernel<<<32, 256, 0, stream>>>(Wn2, wn2bf, (long long)OUT * HID);

    // ---- layer 1 ----
    zero_f32_kernel<<<2048, 256, 0, stream>>>(agg, (long long)N * IN + N);
    sage_scatter_kernel<IN><<<sgrid, IN, 0, stream>>>(xbf, esrc, edst, agg, deg, E);
    neigh_cvt_kernel<<<2048, 256, 0, stream>>>(agg, deg, hneigh, (long long)N * IN);
    sage_gemm_kernel<IN, HID, /*RELU=*/true, /*OUT_BF16=*/true>
        <<<rowTiles, (HID / 16) * 32, 0, stream>>>(
            xbf, hneigh, ws1bf, wn1bf, b1, h1, N);

    // ---- layer 2 (reuse agg; deg unchanged — same graph) ----
    zero_f32_kernel<<<2048, 256, 0, stream>>>(agg, (long long)N * HID);
    sage_scatter_kernel<HID><<<sgrid, HID, 0, stream>>>(h1, esrc, edst, agg, nullptr, E);
    neigh_cvt_kernel<<<2048, 256, 0, stream>>>(agg, deg, hneigh, (long long)N * HID);
    sage_gemm_kernel<HID, OUT, /*RELU=*/false, /*OUT_BF16=*/false>
        <<<rowTiles, (OUT / 16) * 32, 0, stream>>>(
            h1, hneigh, ws2bf, wn2bf, b2, out, N);
}
